// MultiShapelet_Learner_50551765074292
// MI455X (gfx1250) — matrix operand order repaired and checked by reading, not compiled, not run on hardware
//
#include <hip/hip_runtime.h>
#include <hip/hip_bf16.h>

// ---- problem constants (from reference) ----
#define B_BATCH 128
#define Q_LEN   2048
#define C_CH    8
#define L_SHP   256
#define K_LEN   64
#define S_WIN   (Q_LEN - K_LEN + 1)   // 1985 windows
#define KC      (K_LEN * C_CH)        // 512 = GEMM K dim
#define MT      256                   // window tile (M) per block
#define NT      32                    // shapelet tile (N) per block
#define BPAD    (KC + 8)              // LDS row pad: 520 f16 -> conflict-free B reads
#define TSROWS  (MT + K_LEN)          // 320 raw ts rows staged per block

typedef _Float16 f16;
typedef __attribute__((ext_vector_type(16))) _Float16 v16h;
typedef __attribute__((ext_vector_type(8)))  _Float16 v8h;
typedef __attribute__((ext_vector_type(8)))  float    v8f;

// workspace layout (bytes)
#define OFF_WINSQ 0u                            // 128*2048*4 = 1048576
#define OFF_BF16  1048576u                      // 256*512*2  =  262144
#define OFF_SHSQ  (OFF_BF16 + 262144u)          // 256*4 (padded to 4096)
#define OFF_KEYS  (OFF_SHSQ + 4096u)            // 128*256*4  =  131072

// order-preserving float -> uint map (min in key space == min float)
__device__ __forceinline__ unsigned fkey(float f) {
    unsigned u = __float_as_uint(f);
    return (u & 0x80000000u) ? ~u : (u | 0x80000000u);
}
__device__ __forceinline__ float fkey_inv(unsigned k) {
    unsigned u = (k & 0x80000000u) ? (k & 0x7FFFFFFFu) : ~k;
    return __uint_as_float(u);
}

// ---------------------------------------------------------------------------
// 1) per-batch window squared-norms via sliding sum of per-step squares
// ---------------------------------------------------------------------------
__global__ __launch_bounds__(256)
void prep_winsq(const float* __restrict__ ts, float* __restrict__ winsq) {
    __shared__ float ts2[Q_LEN];
    const int b = blockIdx.x, tid = threadIdx.x;
    for (int q = tid; q < Q_LEN; q += 256) {
        const float4* p = (const float4*)(ts + ((size_t)b * Q_LEN + q) * C_CH);
        float4 v0 = p[0], v1 = p[1];
        ts2[q] = v0.x*v0.x + v0.y*v0.y + v0.z*v0.z + v0.w*v0.w
               + v1.x*v1.x + v1.y*v1.y + v1.z*v1.z + v1.w*v1.w;
    }
    __syncthreads();
    for (int w = tid; w < S_WIN; w += 256) {
        float s = 0.f;
        #pragma unroll 8
        for (int k = 0; k < K_LEN; ++k) s += ts2[w + k];
        winsq[(size_t)b * Q_LEN + w] = s;
    }
}

// ---------------------------------------------------------------------------
// 2) shapelets -> f16 [L,512] + sh_sq[l]
// ---------------------------------------------------------------------------
__global__ __launch_bounds__(128)
void prep_shap(const float* __restrict__ shp, f16* __restrict__ Bg,
               float* __restrict__ shsq) {
    __shared__ float red[128];
    const int l = blockIdx.x, tid = threadIdx.x;
    float4 v = *(const float4*)(shp + (size_t)l * KC + tid * 4);
    f16* o = Bg + (size_t)l * KC + tid * 4;
    o[0] = (f16)v.x; o[1] = (f16)v.y; o[2] = (f16)v.z; o[3] = (f16)v.w;
    red[tid] = v.x*v.x + v.y*v.y + v.z*v.z + v.w*v.w;
    __syncthreads();
    for (int off = 64; off > 0; off >>= 1) {
        if (tid < off) red[tid] += red[tid + off];
        __syncthreads();
    }
    if (tid == 0) shsq[l] = red[0];
}

// ---------------------------------------------------------------------------
// 3) init min-keys to "+infinity" (fresh every launch: deterministic)
// ---------------------------------------------------------------------------
__global__ void init_keys(unsigned* __restrict__ keys) {
    keys[(size_t)blockIdx.x * 256 + threadIdx.x] = 0xFFFFFFFFu;
}

// ---------------------------------------------------------------------------
// fragment builders (A from sliding raw-ts rows, B from padded tile)
// A 16x32 f16 layout: lanes 0-15: row M=lane, K 0-7 & 16-23 (time 4t, 4t+2);
//                     lanes 16-31: same rows, K 8-15 & 24-31 (time 4t+1, 4t+3).
// B 32x16 f16 layout: lane column N=lane&15; lanes 0-15 K 0-7 & 16-23, lanes 16-31 K 8-15 & 24-31.
// ---------------------------------------------------------------------------
__device__ __forceinline__ v16h load_a_frag(const f16* tsl, int rlo) {
    v8h lo = *(const v8h*)(tsl + rlo * C_CH);
    v8h hi = *(const v8h*)(tsl + (rlo + 2) * C_CH);
    v16h a;
    #pragma unroll
    for (int i = 0; i < 8; ++i) { a[i] = lo[i]; a[i + 8] = hi[i]; }
    return a;
}
__device__ __forceinline__ v16h load_b_frag(const f16* bp) {
    v8h lo = *(const v8h*)bp;
    v8h hi = *(const v8h*)(bp + 16);
    v16h b;
    #pragma unroll
    for (int i = 0; i < 8; ++i) { b[i] = lo[i]; b[i + 8] = hi[i]; }
    return b;
}

// ---------------------------------------------------------------------------
// 4) WMMA GEMM (cross term) fused with min-pool epilogue.
//    grid = (S tiles=8, L tiles=8, B=128); block = 256 (8 waves).
//    Wave w owns a 32(M) x 32(N) register tile: 2x2 accumulators, so each
//    K-chunk is 8 ds_load_b128 -> 4 v_wmma (2 LDS loads per WMMA).
// ---------------------------------------------------------------------------
__global__ __launch_bounds__(256)
void shapelet_gemm_min(const float* __restrict__ ts,
                       const f16*   __restrict__ Bg,
                       const float* __restrict__ winsq,
                       unsigned*    __restrict__ keys) {
    __shared__ f16   Bl[NT * BPAD];        // 32*520*2 = 33,280 B
    __shared__ f16   tsl[TSROWS * C_CH];   // 320 rows * 8 ch = 5,120 B
    __shared__ float wsq[MT];              // 1,024 B

    const int tid = threadIdx.x;
    const int s0  = blockIdx.x * MT;       // first window of tile
    const int l0  = blockIdx.y * NT;       // first shapelet of tile
    const int b   = blockIdx.z;

    // ---- stage B tile [NT x 512] f16 from preconverted global ----
    for (int i = tid; i < NT * (KC / 8); i += 256) {        // 2048 x uint4
        int n = i >> 6, j = i & 63;
        ((uint4*)&Bl[n * BPAD])[j] =
            ((const uint4*)(Bg + (size_t)(l0 + n) * KC))[j];
    }
    // ---- stage raw ts rows s0..s0+319 (covers 256 windows x 64 steps) ----
    for (int base = tid * 4; base < TSROWS * C_CH; base += 1024) {
        int r = base >> 3, c = base & 7;
        int row = s0 + r;
        float4 v = make_float4(0.f, 0.f, 0.f, 0.f);
        if (row < Q_LEN)
            v = *(const float4*)(ts + ((size_t)b * Q_LEN + row) * C_CH + c);
        tsl[base + 0] = (f16)v.x; tsl[base + 1] = (f16)v.y;
        tsl[base + 2] = (f16)v.z; tsl[base + 3] = (f16)v.w;
    }
    {
        int s = s0 + tid;
        wsq[tid] = (s < S_WIN) ? winsq[(size_t)b * Q_LEN + s] : __builtin_inff();
    }
    __syncthreads();

    const int lane  = tid & 31;
    const int wave  = tid >> 5;
    const int mBase = wave * 32;           // 8 waves cover M=256
    const int lmod  = lane & 15;
    const int lhalf = (lane >> 4) & 1;

    v8f acc[2][2];
    #pragma unroll
    for (int mi = 0; mi < 2; ++mi)
        #pragma unroll
        for (int ni = 0; ni < 2; ++ni)
            acc[mi][ni] = (v8f){0.f,0.f,0.f,0.f,0.f,0.f,0.f,0.f};

    #pragma unroll
    for (int t = 0; t < 16; ++t) {         // K chunks of 32 (4 time steps x 8 ch)
        const int rlo = mBase + lmod + 4 * t + lhalf;       // <= 316; +2+16 <= 334? no: +16+2=334>320? see below
        v16h a0 = load_a_frag(tsl, rlo);                    // rows rlo, rlo+2
        v16h a1 = load_a_frag(tsl, rlo + 16);               // rows rlo+16, rlo+18
        const f16* bp0 = Bl + lmod * BPAD        + 32 * t + 8 * lhalf;
        const f16* bp1 = Bl + (16 + lmod) * BPAD + 32 * t + 8 * lhalf;
        v16h b0 = load_b_frag(bp0);
        v16h b1 = load_b_frag(bp1);
        acc[0][0] = __builtin_amdgcn_wmma_f32_16x16x32_f16(false, a0, false, b0, (short)0, acc[0][0], false, false);
        acc[0][1] = __builtin_amdgcn_wmma_f32_16x16x32_f16(false, a0, false, b1, (short)0, acc[0][1], false, false);
        acc[1][0] = __builtin_amdgcn_wmma_f32_16x16x32_f16(false, a1, false, b0, (short)0, acc[1][0], false, false);
        acc[1][1] = __builtin_amdgcn_wmma_f32_16x16x32_f16(false, a1, false, b1, (short)0, acc[1][1], false, false);
    }
    // max row touched: mBase(224)+lmod(15)+60+lhalf(1)+16+2 = 318 < TSROWS=320. OK.

    // ---- epilogue: per N column, min over this wave's 32 windows ----
    // C/D layout: VGPR r, lanes 0-15 -> M=r ; lanes 16-31 -> M=r+8 ; N = lane&15.
    #pragma unroll
    for (int ni = 0; ni < 2; ++ni) {
        float mn = __builtin_inff();
        #pragma unroll
        for (int mi = 0; mi < 2; ++mi) {
            #pragma unroll
            for (int r = 0; r < 8; ++r) {
                int m = mBase + mi * 16 + r + 8 * lhalf;
                mn = fminf(mn, wsq[m] - 2.0f * acc[mi][ni][r]);
            }
        }
        mn = fminf(mn, __shfl_xor(mn, 16, 32));   // fold lane halves (same N)
        if (lane < 16) {
            int n = l0 + ni * 16 + lmod;
            atomicMin(keys + (size_t)b * L_SHP + n, fkey(mn));
        }
    }
}

// ---------------------------------------------------------------------------
// 5) finalize: decode mins, scale, sigmoid gate, FC -> [B,2]
// ---------------------------------------------------------------------------
__global__ __launch_bounds__(256)
void finalize(const unsigned* __restrict__ keys, const float* __restrict__ shsq,
              const float* __restrict__ gating, const float* __restrict__ w,
              const float* __restrict__ bias, float* __restrict__ out) {
    __shared__ float r0[256], r1[256];
    const int b = blockIdx.x, l = threadIdx.x;          // L == 256 == blockDim
    float val  = fkey_inv(keys[(size_t)b * L_SHP + l]);
    float dist = (val + shsq[l]) * (1.0f / (float)KC);
    float g    = 1.0f / (1.0f + __expf(-gating[l]));    // SATURATION == 1.0
    float v    = dist * g;
    r0[l] = v * w[l];
    r1[l] = v * w[L_SHP + l];
    __syncthreads();
    for (int off = 128; off > 0; off >>= 1) {
        if (l < off) { r0[l] += r0[l + off]; r1[l] += r1[l + off]; }
        __syncthreads();
    }
    if (l == 0) {
        out[b * 2 + 0] = r0[0] + bias[0];
        out[b * 2 + 1] = r1[0] + bias[1];
    }
}

// ---------------------------------------------------------------------------
extern "C" void kernel_launch(void* const* d_in, const int* in_sizes, int n_in,
                              void* d_out, int out_size, void* d_ws, size_t ws_size,
                              hipStream_t stream) {
    const float* ts   = (const float*)d_in[0];   // [128,2048,8]
    const float* shp  = (const float*)d_in[1];   // [256,64,8]
    const float* gat  = (const float*)d_in[2];   // [256]
    const float* fw   = (const float*)d_in[3];   // [2,256]
    const float* fb   = (const float*)d_in[4];   // [2]
    float* out = (float*)d_out;                  // [128,2]

    char* ws = (char*)d_ws;                      // needs ~1.45 MB
    float*    winsq = (float*)   (ws + OFF_WINSQ);
    f16*      Bg    = (f16*)     (ws + OFF_BF16);
    float*    shsq  = (float*)   (ws + OFF_SHSQ);
    unsigned* keys  = (unsigned*)(ws + OFF_KEYS);

    prep_winsq<<<B_BATCH, 256, 0, stream>>>(ts, winsq);
    prep_shap <<<L_SHP,   128, 0, stream>>>(shp, Bg, shsq);
    init_keys <<<(B_BATCH * L_SHP) / 256, 256, 0, stream>>>(keys);

    dim3 grid((S_WIN + MT - 1) / MT, L_SHP / NT, B_BATCH);   // (8, 8, 128)
    shapelet_gemm_min<<<grid, 256, 0, stream>>>(ts, Bg, winsq, keys);

    finalize<<<B_BATCH, 256, 0, stream>>>(keys, shsq, gat, fw, fb, out);
}
